// TransformerBlock_7756710937150
// MI455X (gfx1250) — matrix operand order repaired and checked by reading, not compiled, not run on hardware
//
#include <hip/hip_runtime.h>
#include <hip/hip_bf16.h>
#include <math.h>

// ---------------------------------------------------------------------------
// Swin-style transformer block for MI455X (gfx1250, wave32, WMMA).
// All GEMMs use v_wmma_f32_16x16x32_bf16 (bf16 inputs, f32 accumulate).
// GEMM: 128x128 block tile, double-buffered LDS for load/compute overlap.
// ---------------------------------------------------------------------------

typedef __attribute__((ext_vector_type(16))) __bf16 v16bf;
typedef __attribute__((ext_vector_type(8)))  float  v8f;

union Frag {
    uint4 q[2];      // 32 bytes
    v16bf v;         // 16 x bf16
};

union U4U16 {
    uint4 q;
    unsigned short s[8];
};

__device__ __forceinline__ unsigned short f2b(float f) {
    unsigned int u = __float_as_uint(f);
    unsigned int r = u + 0x7FFFu + ((u >> 16) & 1u);   // round to nearest even
    return (unsigned short)(r >> 16);
}

__device__ __forceinline__ float gelu_f(float v) {
    return 0.5f * v * (1.0f + erff(v * 0.7071067811865475f));
}

// window-order token m -> original token index (same map for gather & scatter)
// Hardcoded H=W=128, WS=8 (matches problem instance; shift offset passed in).
__device__ __forceinline__ int map_token(int m, int off) {
    int win = m >> 6;          // window id
    int t   = m & 63;          // token in window
    int b   = win >> 8;        // 256 windows per image (16x16)
    int wi  = win & 255;
    int wr  = wi >> 4, wc = wi & 15;
    int hs  = wr * 8 + (t >> 3);
    int wsc = wc * 8 + (t & 7);
    int h0  = (hs + off) & 127;
    int w0  = (wsc + off) & 127;
    return (b << 14) | (h0 << 7) | w0;
}

// ---------------------------------------------------------------------------
// f32 -> bf16 conversion (weights)
// ---------------------------------------------------------------------------
__global__ void cvt_kernel(const float* __restrict__ s,
                           unsigned short* __restrict__ d, int n) {
    int i = blockIdx.x * blockDim.x + threadIdx.x;
    if (i < n) d[i] = f2b(s[i]);
}

// ---------------------------------------------------------------------------
// LayerNorm (C=256). One wave per token; 8 waves per block.
// gather!=0: token m reads source row map_token(m, off) (shifted window
// gather) and also writes the gathered f32 row to xcopy (residual source).
// ---------------------------------------------------------------------------
__global__ __launch_bounds__(256)
void ln_kernel(const float* __restrict__ x,
               const float* __restrict__ g, const float* __restrict__ bsh,
               unsigned short* __restrict__ xn, float* __restrict__ xcopy,
               int C, int gather, const int* __restrict__ shiftp) {
    const int lane = threadIdx.x & 31;
    const int wv   = threadIdx.x >> 5;
    const int m    = blockIdx.x * 8 + wv;
    int src = m;
    if (gather) {
        int off = (*shiftp) ? 4 : 0;
        src = map_token(m, off);
    }
    const float* xr = x + (size_t)src * C;
    float v[8];
    float s = 0.f;
#pragma unroll
    for (int i = 0; i < 8; ++i) { v[i] = xr[lane + 32 * i]; s += v[i]; }
#pragma unroll
    for (int d = 16; d > 0; d >>= 1) s += __shfl_xor(s, d, 32);
    float mu = s * (1.0f / 256.0f);
    float qv = 0.f;
#pragma unroll
    for (int i = 0; i < 8; ++i) { float dd = v[i] - mu; qv += dd * dd; }
#pragma unroll
    for (int d = 16; d > 0; d >>= 1) qv += __shfl_xor(qv, d, 32);
    float inv = rsqrtf(qv * (1.0f / 256.0f) + 1e-5f);
#pragma unroll
    for (int i = 0; i < 8; ++i) {
        int c = lane + 32 * i;
        float y = (v[i] - mu) * inv * g[c] + bsh[c];
        xn[(size_t)m * C + c] = f2b(y);
        if (xcopy) xcopy[(size_t)m * C + c] = v[i];
    }
}

// ---------------------------------------------------------------------------
// Generic bf16 GEMM: D[M,N] = A[M,K](bf16,row) * W[N,K](bf16,row)^T + bias
// flags bit0: GELU; bit1: scatter rows of outF through map_token.
// resid (f32, window order, stride N) added if non-null.
// Block: 128x128 tile, 256 threads (8 waves). Wave = 32 rows x 64 cols
// (8 WMMA accumulators). LDS double-buffered: next k-tile global loads are
// issued before compute on the current tile (one barrier per iteration).
// ---------------------------------------------------------------------------
__global__ __launch_bounds__(256)
void gemm_bf16(const unsigned short* __restrict__ A,
               const unsigned short* __restrict__ W,
               const float* __restrict__ bias,
               const float* __restrict__ resid,
               float* __restrict__ outF, unsigned short* __restrict__ outB,
               int M, int N, int K, int flags, const int* __restrict__ shiftp) {
    __shared__ __align__(16) unsigned short As[2][128 * 40];
    __shared__ __align__(16) unsigned short Bs[2][128 * 40];

    const int t    = threadIdx.x;
    const int lane = t & 31;
    const int wv   = t >> 5;
    const int mgrp = wv >> 1;            // 0..3 : 32-row group
    const int ngrp = wv & 1;             // 0..1 : 64-col group
    const int m0   = blockIdx.x * 128;
    const int n0   = blockIdx.y * 128;
    const int lm   = lane & 15;
    const int hi   = lane >> 4;

    const int ldr = t >> 1;              // 0..127 (tile row)
    const int ldc = (t & 1) * 16;        // 0 or 16

    v8f acc[2][4] = {};

    const int nIter = K >> 5;

    // prologue: stage k-tile 0 into buffer 0
    {
        const unsigned short* ap = A + (size_t)(m0 + ldr) * K + ldc;
        const unsigned short* bp = W + (size_t)(n0 + ldr) * K + ldc;
        uint4 ra0 = *(const uint4*)(ap);
        uint4 ra1 = *(const uint4*)(ap + 8);
        uint4 rb0 = *(const uint4*)(bp);
        uint4 rb1 = *(const uint4*)(bp + 8);
        *(uint4*)(&As[0][ldr * 40 + ldc])     = ra0;
        *(uint4*)(&As[0][ldr * 40 + ldc + 8]) = ra1;
        *(uint4*)(&Bs[0][ldr * 40 + ldc])     = rb0;
        *(uint4*)(&Bs[0][ldr * 40 + ldc + 8]) = rb1;
    }
    __syncthreads();

    for (int it = 0; it < nIter; ++it) {
        const int cur = it & 1;
        const bool pre = (it + 1) < nIter;
        uint4 ra0, ra1, rb0, rb1;
        if (pre) {
            const int k0 = (it + 1) << 5;
            const unsigned short* ap = A + (size_t)(m0 + ldr) * K + k0 + ldc;
            const unsigned short* bp = W + (size_t)(n0 + ldr) * K + k0 + ldc;
            ra0 = *(const uint4*)(ap);
            ra1 = *(const uint4*)(ap + 8);
            rb0 = *(const uint4*)(bp);
            rb1 = *(const uint4*)(bp + 8);
        }

        // compute on buffer `cur`
        Frag af[2];
#pragma unroll
        for (int msub = 0; msub < 2; ++msub) {
            const int am = mgrp * 32 + msub * 16 + lm;
            af[msub].q[0] = *(const uint4*)(&As[cur][am * 40 + hi * 8]);
            af[msub].q[1] = *(const uint4*)(&As[cur][am * 40 + 16 + hi * 8]);
        }
#pragma unroll
        for (int nsub = 0; nsub < 4; ++nsub) {
            Frag bf;
            const int bn = ngrp * 64 + nsub * 16 + lm;
            bf.q[0] = *(const uint4*)(&Bs[cur][bn * 40 + hi * 8]);
            bf.q[1] = *(const uint4*)(&Bs[cur][bn * 40 + 16 + hi * 8]);
#pragma unroll
            for (int msub = 0; msub < 2; ++msub) {
                acc[msub][nsub] = __builtin_amdgcn_wmma_f32_16x16x32_bf16(
                    false, af[msub].v, false, bf.v, (short)0,
                    acc[msub][nsub], false, false);
            }
        }

        if (pre) {
            const int nxt = cur ^ 1;
            *(uint4*)(&As[nxt][ldr * 40 + ldc])     = ra0;
            *(uint4*)(&As[nxt][ldr * 40 + ldc + 8]) = ra1;
            *(uint4*)(&Bs[nxt][ldr * 40 + ldc])     = rb0;
            *(uint4*)(&Bs[nxt][ldr * 40 + ldc + 8]) = rb1;
        }
        __syncthreads();
    }

    int off = 0;
    if (flags & 2) off = (*shiftp) ? 4 : 0;

#pragma unroll
    for (int msub = 0; msub < 2; ++msub) {
#pragma unroll
        for (int r = 0; r < 8; ++r) {
            const int gm = m0 + mgrp * 32 + msub * 16 + r + hi * 8;
            size_t orow = (size_t)gm;
            if (flags & 2) orow = (size_t)map_token(gm, off);
#pragma unroll
            for (int nsub = 0; nsub < 4; ++nsub) {
                const int gn = n0 + ngrp * 64 + nsub * 16 + lm;
                float v0 = acc[msub][nsub][r] + bias[gn];
                if (flags & 1) v0 = gelu_f(v0);
                if (resid) v0 += resid[(size_t)gm * N + gn];
                if (outF) outF[orow * N + gn] = v0;
                if (outB) outB[(size_t)gm * N + gn] = f2b(v0);
            }
        }
    }
}

// ---------------------------------------------------------------------------
// Window attention: one block per (window, head). T=64, hd=64.
// S = (Q K^T)/8 via WMMA -> row softmax in LDS -> O = P V via WMMA.
// qkv layout: [M][3C] bf16, q|k|v each C wide, head h at offset h*64.
// ---------------------------------------------------------------------------
__global__ __launch_bounds__(128)
void attn_kernel(const unsigned short* __restrict__ qkv,
                 unsigned short* __restrict__ o, int C3, int C) {
    __shared__ __align__(16) unsigned short Qs[64 * 72];
    __shared__ __align__(16) unsigned short Ks[64 * 72];
    __shared__ __align__(16) unsigned short Vt[64 * 72];   // V transposed [d][k]
    __shared__ __align__(16) float Sf[64 * 65];

    const int win  = blockIdx.x;
    const int hIdx = blockIdx.y;
    const int t    = threadIdx.x;
    const int lane = t & 31;
    const int wv   = t >> 5;
    const int lm   = lane & 15;
    const int hi   = lane >> 4;
    const int base = win * 64;
    const int qoff = hIdx * 64;
    const int koff = C + hIdx * 64;
    const int voff = 2 * C + hIdx * 64;

    // Stage Q, K, V (V transposed). 512 uint4 chunks per matrix, 4/thread.
#pragma unroll
    for (int i = 0; i < 4; ++i) {
        int c    = t + i * 128;
        int row  = c >> 3;
        int col8 = (c & 7) * 8;
        const size_t rb = (size_t)(base + row) * C3;
        *(uint4*)(Qs + row * 72 + col8) = *(const uint4*)(qkv + rb + qoff + col8);
        *(uint4*)(Ks + row * 72 + col8) = *(const uint4*)(qkv + rb + koff + col8);
        U4U16 vv;
        vv.q = *(const uint4*)(qkv + rb + voff + col8);
#pragma unroll
        for (int j = 0; j < 8; ++j) Vt[(col8 + j) * 72 + row] = vv.s[j];
    }
    __syncthreads();

    // ---- S = Q K^T * 0.125 : wave wv owns rows [wv*16, wv*16+16) ----
    const int m0 = wv * 16;
    {
        v8f acc[4] = {v8f{}, v8f{}, v8f{}, v8f{}};
        for (int kk = 0; kk < 64; kk += 32) {
            Frag af;
            const int am = m0 + lm;
            af.q[0] = *(const uint4*)(Qs + am * 72 + kk + hi * 8);
            af.q[1] = *(const uint4*)(Qs + am * 72 + kk + 16 + hi * 8);
#pragma unroll
            for (int j = 0; j < 4; ++j) {
                Frag bf;
                const int bn = j * 16 + lm;
                bf.q[0] = *(const uint4*)(Ks + bn * 72 + kk + hi * 8);
                bf.q[1] = *(const uint4*)(Ks + bn * 72 + kk + 16 + hi * 8);
                acc[j] = __builtin_amdgcn_wmma_f32_16x16x32_bf16(
                    false, af.v, false, bf.v, (short)0, acc[j], false, false);
            }
        }
#pragma unroll
        for (int j = 0; j < 4; ++j)
#pragma unroll
            for (int r = 0; r < 8; ++r)
                Sf[(m0 + r + hi * 8) * 65 + j * 16 + lm] = acc[j][r] * 0.125f;
    }
    __syncthreads();

    // ---- row softmax, write P (bf16) over Qs ----
    unsigned short* Ps = Qs;
    if (t < 64) {
        float mx = -3.4e38f;
#pragma unroll
        for (int n = 0; n < 64; ++n) mx = fmaxf(mx, Sf[t * 65 + n]);
        float sum = 0.f;
#pragma unroll
        for (int n = 0; n < 64; ++n) {
            float e = __expf(Sf[t * 65 + n] - mx);
            Sf[t * 65 + n] = e;
            sum += e;
        }
        float inv = 1.0f / sum;
#pragma unroll
        for (int n = 0; n < 64; ++n)
            Ps[t * 72 + n] = f2b(Sf[t * 65 + n] * inv);
    }
    __syncthreads();

    // ---- O = P V ----
    {
        v8f acc[4] = {v8f{}, v8f{}, v8f{}, v8f{}};
        for (int kk = 0; kk < 64; kk += 32) {
            Frag af;
            const int am = m0 + lm;
            af.q[0] = *(const uint4*)(Ps + am * 72 + kk + hi * 8);
            af.q[1] = *(const uint4*)(Ps + am * 72 + kk + 16 + hi * 8);
#pragma unroll
            for (int j = 0; j < 4; ++j) {
                Frag bf;
                const int bd = j * 16 + lm;       // output dim d
                bf.q[0] = *(const uint4*)(Vt + bd * 72 + kk + hi * 8);
                bf.q[1] = *(const uint4*)(Vt + bd * 72 + kk + 16 + hi * 8);
                acc[j] = __builtin_amdgcn_wmma_f32_16x16x32_bf16(
                    false, af.v, false, bf.v, (short)0, acc[j], false, false);
            }
        }
#pragma unroll
        for (int j = 0; j < 4; ++j)
#pragma unroll
            for (int r = 0; r < 8; ++r) {
                int gRow = base + m0 + r + hi * 8;
                int gCol = hIdx * 64 + j * 16 + lm;
                o[(size_t)gRow * C + gCol] = f2b(acc[j][r]);
            }
    }
}

// ---------------------------------------------------------------------------
// host launcher
// ---------------------------------------------------------------------------
extern "C" void kernel_launch(void* const* d_in, const int* in_sizes, int n_in,
                              void* d_out, int out_size, void* d_ws, size_t ws_size,
                              hipStream_t stream) {
    (void)n_in; (void)out_size; (void)ws_size;
    const float* x    = (const float*)d_in[0];
    const float* ln1g = (const float*)d_in[1];
    const float* ln1b = (const float*)d_in[2];
    const float* wqkv = (const float*)d_in[3];
    const float* bqkv = (const float*)d_in[4];
    const float* wo   = (const float*)d_in[5];
    const float* bo   = (const float*)d_in[6];
    const float* ln2g = (const float*)d_in[7];
    const float* ln2b = (const float*)d_in[8];
    const float* w1   = (const float*)d_in[9];
    const float* b1   = (const float*)d_in[10];
    const float* w2   = (const float*)d_in[11];
    const float* b2   = (const float*)d_in[12];
    const int* shiftp = (const int*)d_in[15];

    const int C  = in_sizes[1];          // 256
    const int Cm = in_sizes[10];         // 1024
    const int C3 = 3 * C;                // 768
    const int M  = in_sizes[0] / C;      // 131072 tokens

    // workspace carve-up (256-byte aligned chunks)
    char* p = (char*)d_ws;
    auto carve = [&](size_t bytes) -> char* {
        char* r = p;
        p += (bytes + 255) & ~(size_t)255;
        return r;
    };
    unsigned short* wqkv_b = (unsigned short*)carve((size_t)C3 * C * 2);
    unsigned short* wo_b   = (unsigned short*)carve((size_t)C * C * 2);
    unsigned short* w1_b   = (unsigned short*)carve((size_t)Cm * C * 2);
    unsigned short* w2_b   = (unsigned short*)carve((size_t)C * Cm * 2);
    float*          xw     = (float*)carve((size_t)M * C * 4);   // gathered x (residual 1)
    unsigned short* xn     = (unsigned short*)carve((size_t)M * C * 2);  // ln1 out / ln2 out
    unsigned short* qkvb   = (unsigned short*)carve((size_t)M * C3 * 2);
    unsigned short* ob     = (unsigned short*)carve((size_t)M * C * 2);  // attn out
    float*          x1     = (float*)carve((size_t)M * C * 4);   // post-attn residual
    unsigned short* hm     = (unsigned short*)carve((size_t)M * Cm * 2); // mlp mid

    // 1. weights -> bf16
    {
        int n;
        n = C3 * C; cvt_kernel<<<(n + 255) / 256, 256, 0, stream>>>(wqkv, wqkv_b, n);
        n = C * C;  cvt_kernel<<<(n + 255) / 256, 256, 0, stream>>>(wo,   wo_b,   n);
        n = Cm * C; cvt_kernel<<<(n + 255) / 256, 256, 0, stream>>>(w1,   w1_b,   n);
        n = C * Cm; cvt_kernel<<<(n + 255) / 256, 256, 0, stream>>>(w2,   w2_b,   n);
    }

    // 2. LN1 + shifted-window gather (also copies gathered x f32)
    ln_kernel<<<M / 8, 256, 0, stream>>>(x, ln1g, ln1b, xn, xw, C, 1, shiftp);

    // 3. QKV GEMM: [M,C] x [C3,C]^T -> bf16
    gemm_bf16<<<dim3(M / 128, C3 / 128), 256, 0, stream>>>(
        xn, wqkv_b, bqkv, nullptr, nullptr, qkvb, M, C3, C, 0, nullptr);

    // 4. windowed attention (2048 windows x 4 heads)
    attn_kernel<<<dim3(M / 64, 4), 128, 0, stream>>>(qkvb, ob, C3, C);

    // 5. output projection + residual (f32, window order)
    gemm_bf16<<<dim3(M / 128, C / 128), 256, 0, stream>>>(
        ob, wo_b, bo, xw, x1, nullptr, M, C, C, 0, nullptr);

    // 6. LN2 (no gather)
    ln_kernel<<<M / 8, 256, 0, stream>>>(x1, ln2g, ln2b, xn, nullptr, C, 0, nullptr);

    // 7. MLP up + exact GELU -> bf16
    gemm_bf16<<<dim3(M / 128, Cm / 128), 256, 0, stream>>>(
        xn, w1_b, b1, nullptr, nullptr, hm, M, Cm, C, 1, nullptr);

    // 8. MLP down + residual + un-window/un-shift scatter to d_out (f32)
    gemm_bf16<<<dim3(M / 128, C / 128), 256, 0, stream>>>(
        hm, w2_b, b2, x1, (float*)d_out, nullptr, M, C, Cm, 2, shiftp);
}
